// GNNConvLayer_85933705658978
// MI455X (gfx1250) — matrix operand order repaired and checked by reading, not compiled, not run on hardware
//
#include <hip/hip_runtime.h>

// Problem constants (match reference)
#define BB 2
#define NN 50000
#define CC 128
#define HH 512
#define EE 800000

typedef __attribute__((ext_vector_type(16))) __bf16 v16bf;
typedef __attribute__((ext_vector_type(8)))  __bf16 v8bf;
typedef __attribute__((ext_vector_type(8)))  float  v8f;

__device__ __forceinline__ v16bf cat8(v8bf lo, v8bf hi) {
  v16bf r;
#pragma unroll
  for (int i = 0; i < 8; ++i) { r[i] = lo[i]; r[i + 8] = hi[i]; }
  return r;
}

__device__ __forceinline__ void fadd_atomic(float* p, float v) {
  // Relaxed, device-scope, no return value -> single global_atomic_add_f32
  // (non-returning RMW, TH[0]=0), resolved in L2 where the 25.6 MB/batch
  // node array is resident.
  __hip_atomic_fetch_add(p, v, __ATOMIC_RELAXED, __HIP_MEMORY_SCOPE_AGENT);
}

// ---------------- scatter-add: agg[b,dst,:] += x[b,src,:] ----------------
// one wave32 per (batch, edge); each lane moves 4 channels (float4 gather).
__global__ __launch_bounds__(256) void gin_scatter(
    const float* __restrict__ x, const int* __restrict__ ei,
    float* __restrict__ agg)
{
  int gt   = blockIdx.x * 256 + threadIdx.x;
  int lane = gt & 31;
  int slot = gt >> 5;                      // (b, e)
  if (slot >= BB * EE) return;
  int e = slot % EE;
  int b = slot / EE;
  int src = ei[e];
  int dst = ei[EE + e];
  const float4 v = ((const float4*)(x + ((size_t)b * NN + src) * CC))[lane];
  float* a = agg + ((size_t)b * NN + dst) * CC + (size_t)lane * 4;
  fadd_atomic(a + 0, v.x);
  fadd_atomic(a + 1, v.y);
  fadd_atomic(a + 2, v.z);
  fadd_atomic(a + 3, v.w);
}

// ------------- pack weight K x N (row-major f32) -> col-major bf16 -------------
// Wp[n*K + k] = bf16(W[k*N + n]); B-fragment then reads 16 consecutive K per lane.
__global__ __launch_bounds__(256) void pack_w(
    const float* __restrict__ W, __bf16* __restrict__ Wp, int K, int N)
{
  int t = blockIdx.x * 256 + threadIdx.x;
  if (t >= K * N) return;
  int n = t / K, k = t % K;
  Wp[t] = (__bf16)W[(size_t)k * N + n];
}

// ---------------- fused GIN MLP for 32 rows per workgroup ----------------
// out = relu((x+agg) @ W1 + b1) @ W2 + b2, h1 kept in LDS (bf16).
#define SAP 136   // padded LDS stride for A  (128 + 8 bf16 -> +4 banks/row)
#define SHP 520   // padded LDS stride for h1 (512 + 8 bf16 -> +4 banks/row)

__global__ __launch_bounds__(256) void gin_fused_mlp(
    const float* __restrict__ x, const float* __restrict__ agg,
    const __bf16* __restrict__ W1p, const float* __restrict__ b1,
    const __bf16* __restrict__ W2p, const float* __restrict__ b2,
    float* __restrict__ out)
{
  __shared__ __bf16 sA[32 * SAP];   // ~8.5 KB : (x+agg) tile, bf16
  __shared__ __bf16 sH[32 * SHP];   // ~32.5 KB: relu(h1) tile, bf16

  const int t    = threadIdx.x;
  const int wave = t >> 5;
  const int lane = t & 31;
  const int half = lane >> 4;       // which half-wave (K-offset select)
  const int ln   = lane & 15;       // row/col within tile
  const size_t row0 = (size_t)blockIdx.x * 32;

  // ---- stage A = x + agg -> bf16 LDS (32 x 128) ----
  {
    int r  = t >> 3;                // 0..31
    int c0 = (t & 7) * 16;          // 0..112
    const float4* xp = (const float4*)(x   + (row0 + r) * CC + c0);
    const float4* ap = (const float4*)(agg + (row0 + r) * CC + c0);
    __bf16* sp = sA + r * SAP + c0;
#pragma unroll
    for (int i = 0; i < 4; ++i) {
      float4 xv = xp[i], av = ap[i];
      sp[i * 4 + 0] = (__bf16)(xv.x + av.x);
      sp[i * 4 + 1] = (__bf16)(xv.y + av.y);
      sp[i * 4 + 2] = (__bf16)(xv.z + av.z);
      sp[i * 4 + 3] = (__bf16)(xv.w + av.w);
    }
  }
  __syncthreads();

  // ---- GEMM1: (32x128) @ (128x512), wave tile = 16 rows x 128 cols ----
  {
    const int rs = (wave & 1) * 16;        // row strip
    const int cc = (wave >> 1) * 128;      // col chunk (8 tiles of 16)
    v8f acc[8];
#pragma unroll
    for (int i = 0; i < 8; ++i) acc[i] = (v8f){};

#pragma unroll
    for (int kk = 0; kk < CC; kk += 32) {
      // A fragment: row = rs+ln, K runs [kk+16*half, +16)
      const __bf16* ab = sA + (rs + ln) * SAP + kk + 16 * half;
      v16bf afrag = cat8(*(const v8bf*)ab, *(const v8bf*)(ab + 8));
#pragma unroll
      for (int tc = 0; tc < 8; ++tc) {
        int n = cc + tc * 16 + ln;
        const __bf16* bb = W1p + (size_t)n * CC + kk + 16 * half;
        v16bf bfrag = cat8(*(const v8bf*)bb, *(const v8bf*)(bb + 8));
        acc[tc] = __builtin_amdgcn_wmma_f32_16x16x32_bf16(
            false, afrag, false, bfrag, (short)0, acc[tc], false, false);
      }
    }

    // bias + relu -> sH (bf16)
#pragma unroll
    for (int tc = 0; tc < 8; ++tc) {
      int n = cc + tc * 16 + ln;
      float bv = b1[n];
#pragma unroll
      for (int i = 0; i < 8; ++i) {
        int r = rs + i + 8 * half;           // C/D layout: VGPR i -> row i (+8 for hi half)
        float v = acc[tc][i] + bv;
        sH[r * SHP + n] = (__bf16)(v > 0.f ? v : 0.f);
      }
    }
  }
  __syncthreads();

  // ---- GEMM2: (32x512) @ (512x128), wave tile = 16 rows x 32 cols ----
  {
    const int rs = (wave & 1) * 16;
    const int cc = (wave >> 1) * 32;       // 2 tiles of 16
    v8f acc[2];
    acc[0] = (v8f){};
    acc[1] = (v8f){};

#pragma unroll
    for (int kk = 0; kk < HH; kk += 32) {
      const __bf16* ab = sH + (rs + ln) * SHP + kk + 16 * half;
      v16bf afrag = cat8(*(const v8bf*)ab, *(const v8bf*)(ab + 8));
#pragma unroll
      for (int tc = 0; tc < 2; ++tc) {
        int n = cc + tc * 16 + ln;
        const __bf16* bb = W2p + (size_t)n * HH + kk + 16 * half;
        v16bf bfrag = cat8(*(const v8bf*)bb, *(const v8bf*)(bb + 8));
        acc[tc] = __builtin_amdgcn_wmma_f32_16x16x32_bf16(
            false, afrag, false, bfrag, (short)0, acc[tc], false, false);
      }
    }

#pragma unroll
    for (int tc = 0; tc < 2; ++tc) {
      int n = cc + tc * 16 + ln;
      float bv = b2[n];
#pragma unroll
      for (int i = 0; i < 8; ++i) {
        size_t r = row0 + rs + i + 8 * half;
        out[r * CC + n] = acc[tc][i] + bv;
      }
    }
  }
}

extern "C" void kernel_launch(void* const* d_in, const int* in_sizes, int n_in,
                              void* d_out, int out_size, void* d_ws, size_t ws_size,
                              hipStream_t stream) {
  const float* x0   = (const float*)d_in[0];
  const int*   ei   = (const int*)d_in[1];   // edge_index (2, E)
  const float* W1_0 = (const float*)d_in[2];
  const float* b1_0 = (const float*)d_in[3];
  const float* W2_0 = (const float*)d_in[4];
  const float* b2_0 = (const float*)d_in[5];
  const float* W1_1 = (const float*)d_in[6];
  const float* b1_1 = (const float*)d_in[7];
  const float* W2_1 = (const float*)d_in[8];
  const float* b2_1 = (const float*)d_in[9];

  const size_t BNC = (size_t)BB * NN * CC;
  float*  agg  = (float*)d_ws;               // 51.2 MB
  float*  xmid = agg + BNC;                  // 51.2 MB
  __bf16* w1p  = (__bf16*)(xmid + BNC);      // 128 KB
  __bf16* w2p  = w1p + (size_t)CC * HH;      // 128 KB

  dim3 blk(256);
  const int packBlocks = (CC * HH + 255) / 256;
  const int scatBlocks = (BB * EE * 32) / 256;
  const int gemmBlocks = (BB * NN) / 32;

  // ---- layer 0 ----
  pack_w<<<packBlocks, blk, 0, stream>>>(W1_0, w1p, CC, HH);
  pack_w<<<packBlocks, blk, 0, stream>>>(W2_0, w2p, HH, CC);
  hipMemsetAsync(agg, 0, BNC * sizeof(float), stream);
  gin_scatter<<<scatBlocks, blk, 0, stream>>>(x0, ei, agg);
  gin_fused_mlp<<<gemmBlocks, blk, 0, stream>>>(x0, agg, w1p, b1_0, w2p, b2_0, xmid);

  // ---- layer 1 ----
  pack_w<<<packBlocks, blk, 0, stream>>>(W1_1, w1p, CC, HH);
  pack_w<<<packBlocks, blk, 0, stream>>>(W2_1, w2p, HH, CC);
  hipMemsetAsync(agg, 0, BNC * sizeof(float), stream);
  gin_scatter<<<scatBlocks, blk, 0, stream>>>(xmid, ei, agg);
  gin_fused_mlp<<<gemmBlocks, blk, 0, stream>>>(xmid, agg, w1p, b1_1, w2p, b2_1, (float*)d_out);
}
